// FusedSMOE_15925738733961
// MI455X (gfx1250) — matrix operand order repaired and checked by reading, not compiled, not run on hardware
//
#include <hip/hip_runtime.h>
#include <stdint.h>

#define DIM   1024
#define DFF   2048
#define DHALF 1024
#define NEXP  8

typedef __bf16 bf16_t;
typedef bf16_t v16bf __attribute__((ext_vector_type(16)));
typedef unsigned short v16us __attribute__((ext_vector_type(16)));
typedef float v8f __attribute__((ext_vector_type(8)));
typedef unsigned int v4u __attribute__((ext_vector_type(4)));
typedef float v4f __attribute__((ext_vector_type(4)));

typedef __attribute__((address_space(3))) unsigned short lds_us;

union Frag32B {
  v4u  q[2];
  v16us us;
  v16bf bf;
};

static __device__ __forceinline__ unsigned short f32_to_bf16(float f) {
  unsigned int u = __builtin_bit_cast(unsigned int, f);
  u += 0x7FFFu + ((u >> 16) & 1u);   // round-to-nearest-even
  return (unsigned short)(u >> 16);
}

static __device__ __forceinline__ v8f vzero8() {
  v8f z;
#pragma unroll
  for (int i = 0; i < 8; ++i) z[i] = 0.0f;
  return z;
}

// One instruction: each lane copies 16B global -> LDS, tracked by ASYNCcnt.
static __device__ __forceinline__ void async_ld16(void* ldsdst, const void* gsrc) {
  asm volatile("global_load_async_to_lds_b128 %0, %1, off"
               :: "v"((lds_us*)ldsdst), "v"(gsrc)
               : "memory");
}

// ---------------- fp32 -> bf16 bulk convert (weights) ----------------
__global__ __launch_bounds__(256) void moe_cvt_bf16(const float* __restrict__ src,
                                                    unsigned short* __restrict__ dst,
                                                    long long n) {
  long long i = ((long long)blockIdx.x * 256 + threadIdx.x) * 8;
  if (i + 8 > n) return;
  const v4f* s = (const v4f*)(src + i);
  v4f a = s[0], b = s[1];
  v4u o;
  o[0] = (unsigned)f32_to_bf16(a[0]) | ((unsigned)f32_to_bf16(a[1]) << 16);
  o[1] = (unsigned)f32_to_bf16(a[2]) | ((unsigned)f32_to_bf16(a[3]) << 16);
  o[2] = (unsigned)f32_to_bf16(b[0]) | ((unsigned)f32_to_bf16(b[1]) << 16);
  o[3] = (unsigned)f32_to_bf16(b[2]) | ((unsigned)f32_to_bf16(b[3]) << 16);
  *(v4u*)(dst + i) = o;
}

// -------- gating: logits -> softmax -> top2 -> combine + per-expert token lists --------
__global__ __launch_bounds__(256) void moe_gate(const float* __restrict__ x,
                                                const float* __restrict__ Wg,
                                                const float* __restrict__ bg,
                                                unsigned short* __restrict__ xb,
                                                float* __restrict__ combine,
                                                int* __restrict__ cnt,
                                                int* __restrict__ list,
                                                int lstride) {
  __shared__ float wgs[NEXP * DIM];
  __shared__ float bgs[NEXP];
  const int tid = threadIdx.x;
  for (int i = tid; i < NEXP * DIM; i += 256) wgs[i] = Wg[i];
  if (tid < NEXP) bgs[tid] = bg[tid];
  __syncthreads();

  const int lane = tid & 31;
  const int tok = blockIdx.x * 8 + (tid >> 5);   // one wave32 per token
  const float* xr = x + (size_t)tok * DIM;

  float acc[NEXP];
#pragma unroll
  for (int e = 0; e < NEXP; ++e) acc[e] = 0.0f;

  for (int i = 0; i < DIM / 32; ++i) {
    int d = i * 32 + lane;
    float xv = xr[d];
    xb[(size_t)tok * DIM + d] = f32_to_bf16(xv);
#pragma unroll
    for (int e = 0; e < NEXP; ++e) acc[e] += xv * wgs[e * DIM + d];
  }
#pragma unroll
  for (int e = 0; e < NEXP; ++e) {
    float v = acc[e];
#pragma unroll
    for (int m = 16; m >= 1; m >>= 1) v += __shfl_xor(v, m, 32);
    acc[e] = v;
  }
  if (lane == 0) {
    float mx = -3.4e38f;
#pragma unroll
    for (int e = 0; e < NEXP; ++e) { acc[e] += bgs[e]; mx = fmaxf(mx, acc[e]); }
    float p[NEXP];
#pragma unroll
    for (int e = 0; e < NEXP; ++e) p[e] = __expf(acc[e] - mx);
    int i1 = 0;
#pragma unroll
    for (int e = 1; e < NEXP; ++e) if (p[e] > p[i1]) i1 = e;
    int i2 = (i1 == 0) ? 1 : 0;
#pragma unroll
    for (int e = 0; e < NEXP; ++e) if (e != i1 && p[e] > p[i2]) i2 = e;
    float s = p[i1] + p[i2];
    float wa = p[i1] / s, wb = p[i2] / s;
#pragma unroll
    for (int e = 0; e < NEXP; ++e)
      combine[(size_t)tok * NEXP + e] = (e == i1) ? wa : ((e == i2) ? wb : 0.0f);
    int p1 = atomicAdd(&cnt[i1], 1);
    list[(size_t)i1 * lstride + p1] = tok;
    int p2 = atomicAdd(&cnt[i2], 1);
    list[(size_t)i2 * lstride + p2] = tok;
  }
}

// -------- FFN1 (gathered): act[pos,:] = silu(x[tok]@Wg.T) * (x[tok]@Wu.T) --------
// 256 thr (8 waves); tile M=128 gathered tokens x N=64; K=1024 in 32-steps.
// 4-deep LDS ring fed by global_load_async_to_lds_b128 (ASYNCcnt).
__global__ __launch_bounds__(256) void moe_ffn1(const unsigned short* __restrict__ xb,
                                                const unsigned short* __restrict__ w1b,
                                                unsigned short* __restrict__ act,
                                                const int* __restrict__ cnt,
                                                const int* __restrict__ list,
                                                int lstride, int e) {
  const int nTok = cnt[e];
  const int tiles = (nTok + 127) >> 7;
  if ((int)blockIdx.y >= tiles) return;

  __shared__ __align__(16) unsigned short Xs[4][128 * 32];
  __shared__ __align__(16) unsigned short Wgs[4][64 * 32];
  __shared__ __align__(16) unsigned short Wus[4][64 * 32];

  const int tid  = threadIdx.x;
  const int lane = tid & 31;
  const int wave = tid >> 5;
  const int wm = (wave & 3) * 32;
  const int wn = (wave >> 2) * 32;
  const int posBase = blockIdx.y * 128;
  const int nBase   = blockIdx.x * 64;
  const unsigned short* w1e = w1b + (size_t)e * DFF * DIM;
  const int* le = list + (size_t)e * lstride + posBase;

  const int n16 = lane & 15;
  const int hh  = lane >> 4;
  const int sm = tid >> 2;
  const int sq = (tid & 3) * 8;

  // gathered source rows for this thread's staging slots (padding entries are 0)
  const int tokA = le[sm];
  const int tokB = le[sm + 64];

  v8f accg[2][2], accu[2][2];
#pragma unroll
  for (int i = 0; i < 2; ++i)
#pragma unroll
    for (int j = 0; j < 2; ++j) { accg[i][j] = vzero8(); accu[i][j] = vzero8(); }

  auto stage = [&](int buf, int k0) {
    async_ld16(&Xs[buf][sm * 32 + sq],        xb  + (size_t)tokA                 * DIM + k0 + sq);
    async_ld16(&Xs[buf][(sm + 64) * 32 + sq], xb  + (size_t)tokB                 * DIM + k0 + sq);
    async_ld16(&Wgs[buf][sm * 32 + sq],       w1e + (size_t)(nBase + sm)         * DIM + k0 + sq);
    async_ld16(&Wus[buf][sm * 32 + sq],       w1e + (size_t)(DHALF + nBase + sm) * DIM + k0 + sq);
  };

  stage(0, 0);
#pragma unroll 4
  for (int i = 0; i < DIM / 32; ++i) {
    const int cur = i & 3;
    if (i + 1 < DIM / 32) {
      stage((i + 1) & 3, (i + 1) * 32);
      asm volatile("s_wait_asynccnt 4" ::: "memory");   // current buffer's copies landed
    } else {
      asm volatile("s_wait_asynccnt 0" ::: "memory");
    }
    __syncthreads();

    Frag32B a[2], bgf[2], buf_[2];
#pragma unroll
    for (int mi = 0; mi < 2; ++mi) {
      const v4u* xrow = (const v4u*)(Xs[cur] + (wm + mi * 16 + n16) * 32);
      a[mi].q[0] = xrow[2 * hh];
      a[mi].q[1] = xrow[2 * hh + 1];
    }
#pragma unroll
    for (int ni = 0; ni < 2; ++ni) {
      const v4u* grow = (const v4u*)(Wgs[cur] + (wn + ni * 16 + n16) * 32);
      bgf[ni].q[0] = grow[2 * hh];
      bgf[ni].q[1] = grow[2 * hh + 1];
      const v4u* urow = (const v4u*)(Wus[cur] + (wn + ni * 16 + n16) * 32);
      buf_[ni].q[0] = urow[2 * hh];
      buf_[ni].q[1] = urow[2 * hh + 1];
    }
#pragma unroll
    for (int mi = 0; mi < 2; ++mi)
#pragma unroll
      for (int ni = 0; ni < 2; ++ni) {
        accg[mi][ni] = __builtin_amdgcn_wmma_f32_16x16x32_bf16(
            false, a[mi].bf, false, bgf[ni].bf, (short)0, accg[mi][ni], false, false);
        accu[mi][ni] = __builtin_amdgcn_wmma_f32_16x16x32_bf16(
            false, a[mi].bf, false, buf_[ni].bf, (short)0, accu[mi][ni], false, false);
      }
  }

  // SwiGLU epilogue -> bf16 act (indexed by list position)
#pragma unroll
  for (int mi = 0; mi < 2; ++mi)
#pragma unroll
    for (int ni = 0; ni < 2; ++ni) {
      v8f g = accg[mi][ni], u = accu[mi][ni];
      int col = nBase + wn + ni * 16 + n16;
#pragma unroll
      for (int r = 0; r < 8; ++r) {
        int row = posBase + wm + mi * 16 + r + 8 * hh;
        float gv = g[r];
        float av = (gv / (1.0f + __expf(-gv))) * u[r];
        act[(size_t)row * DHALF + col] = f32_to_bf16(av);
      }
    }
}

// -------- FFN2 (gathered): out[tok,:] += combine[tok,e] * (act[pos,:] @ w2e.T) --------
__global__ __launch_bounds__(256) void moe_ffn2(const unsigned short* __restrict__ act,
                                                const unsigned short* __restrict__ w2b,
                                                const float* __restrict__ combine,
                                                float* __restrict__ out,
                                                const int* __restrict__ cnt,
                                                const int* __restrict__ list,
                                                int lstride, int e) {
  const int nTok = cnt[e];
  const int tiles = (nTok + 127) >> 7;
  if ((int)blockIdx.y >= tiles) return;

  __shared__ __align__(16) unsigned short As[4][128 * 32];
  __shared__ __align__(16) unsigned short Ws[4][64 * 32];
  __shared__ int   Ls[128];
  __shared__ float cs[128];

  const int tid  = threadIdx.x;
  const int lane = tid & 31;
  const int wave = tid >> 5;
  const int wm = (wave & 3) * 32;
  const int wn = (wave >> 2) * 32;
  const int posBase = blockIdx.y * 128;
  const int nBase   = blockIdx.x * 64;
  const unsigned short* w2e = w2b + (size_t)e * DIM * DHALF;

  const int n16 = lane & 15;
  const int hh  = lane >> 4;
  const int sm = tid >> 2;
  const int sq = (tid & 3) * 8;

  if (tid < 128) {
    int pos = posBase + tid;
    int tk = (pos < nTok) ? list[(size_t)e * lstride + pos] : 0;
    Ls[tid] = tk;
    cs[tid] = (pos < nTok) ? combine[(size_t)tk * NEXP + e] : 0.0f;
  }

  v8f acc[2][2];
#pragma unroll
  for (int i = 0; i < 2; ++i)
#pragma unroll
    for (int j = 0; j < 2; ++j) acc[i][j] = vzero8();

  auto stage = [&](int buf, int k0) {
    async_ld16(&As[buf][sm * 32 + sq],        act + (size_t)(posBase + sm)      * DHALF + k0 + sq);
    async_ld16(&As[buf][(sm + 64) * 32 + sq], act + (size_t)(posBase + sm + 64) * DHALF + k0 + sq);
    async_ld16(&Ws[buf][sm * 32 + sq],        w2e + (size_t)(nBase + sm)        * DHALF + k0 + sq);
  };

  stage(0, 0);
#pragma unroll 4
  for (int i = 0; i < DHALF / 32; ++i) {
    const int cur = i & 3;
    if (i + 1 < DHALF / 32) {
      stage((i + 1) & 3, (i + 1) * 32);
      asm volatile("s_wait_asynccnt 3" ::: "memory");
    } else {
      asm volatile("s_wait_asynccnt 0" ::: "memory");
    }
    __syncthreads();

    Frag32B a[2], b[2];
#pragma unroll
    for (int mi = 0; mi < 2; ++mi) {
      const v4u* arow = (const v4u*)(As[cur] + (wm + mi * 16 + n16) * 32);
      a[mi].q[0] = arow[2 * hh];
      a[mi].q[1] = arow[2 * hh + 1];
    }
#pragma unroll
    for (int ni = 0; ni < 2; ++ni) {
      const v4u* wrow = (const v4u*)(Ws[cur] + (wn + ni * 16 + n16) * 32);
      b[ni].q[0] = wrow[2 * hh];
      b[ni].q[1] = wrow[2 * hh + 1];
    }
#pragma unroll
    for (int mi = 0; mi < 2; ++mi)
#pragma unroll
      for (int ni = 0; ni < 2; ++ni)
        acc[mi][ni] = __builtin_amdgcn_wmma_f32_16x16x32_bf16(
            false, a[mi].bf, false, b[ni].bf, (short)0, acc[mi][ni], false, false);
  }

#pragma unroll
  for (int mi = 0; mi < 2; ++mi)
#pragma unroll
    for (int ni = 0; ni < 2; ++ni) {
      int col = nBase + wn + ni * 16 + n16;
#pragma unroll
      for (int r = 0; r < 8; ++r) {
        int lrow = wm + mi * 16 + r + 8 * hh;
        if (posBase + lrow < nTok) {
          int row = Ls[lrow];
          out[(size_t)row * DIM + col] += acc[mi][ni][r] * cs[lrow];
        }
      }
    }
}

extern "C" void kernel_launch(void* const* d_in, const int* in_sizes, int n_in,
                              void* d_out, int out_size, void* d_ws, size_t ws_size,
                              hipStream_t stream) {
  (void)n_in; (void)ws_size;
  const float* x  = (const float*)d_in[0];
  const float* Wg = (const float*)d_in[1];
  const float* bg = (const float*)d_in[2];
  const float* w1 = (const float*)d_in[3];
  const float* w2 = (const float*)d_in[4];
  float* out = (float*)d_out;

  const long long T    = (long long)in_sizes[0] / DIM;   // 8192 tokens
  const long long n_w1 = (long long)NEXP * DFF * DIM;
  const long long n_w2 = (long long)NEXP * DIM * DHALF;
  const int lstride    = (int)T + 128;                   // padded per-expert list stride

  char* ws = (char*)d_ws;
  size_t off = 0;
  auto carve = [&](size_t bytes) -> char* {
    off = (off + 255) & ~((size_t)255);
    char* p = ws + off;
    off += bytes;
    return p;
  };
  unsigned short* xb      = (unsigned short*)carve((size_t)T * DIM * 2);
  unsigned short* w1b     = (unsigned short*)carve((size_t)n_w1 * 2);
  unsigned short* w2b     = (unsigned short*)carve((size_t)n_w2 * 2);
  unsigned short* actb    = (unsigned short*)carve(((size_t)T + 128) * DHALF * 2);
  float*          combine = (float*)carve((size_t)T * NEXP * 4);
  int*            cnt     = (int*)carve((size_t)NEXP * 4);
  int*            list    = (int*)carve((size_t)NEXP * lstride * 4);

  // re-zero routing state every call (graph-capture legal, keeps launch deterministic)
  hipMemsetAsync(cnt, 0, (size_t)NEXP * 4, stream);
  hipMemsetAsync(list, 0, (size_t)NEXP * lstride * 4, stream);
  hipMemsetAsync(d_out, 0, (size_t)out_size * sizeof(float), stream);

  moe_cvt_bf16<<<(unsigned)(n_w1 / (8 * 256)), 256, 0, stream>>>(w1, w1b, n_w1);
  moe_cvt_bf16<<<(unsigned)(n_w2 / (8 * 256)), 256, 0, stream>>>(w2, w2b, n_w2);
  moe_gate<<<(unsigned)(T / 8), 256, 0, stream>>>(x, Wg, bg, xb, combine, cnt, list, lstride);

  dim3 g1(DHALF / 64, (unsigned)(T / 128));   // max tiles; blocks early-exit past cnt[e]
  dim3 g2(DIM / 64, (unsigned)(T / 128));
  for (int e = 0; e < NEXP; ++e) {
    moe_ffn1<<<g1, 256, 0, stream>>>(xb, w1b, actb, cnt, list, lstride, e);
    moe_ffn2<<<g2, 256, 0, stream>>>(actb, w2b, combine, out, cnt, list, lstride, e);
  }
}
